// Attention_55070070670307
// MI455X (gfx1250) — matrix operand order repaired and checked by reading, not compiled, not run on hardware
//
#include <hip/hip_runtime.h>

#define HEADS  8
#define DIMH   64
#define SEQ    4096
#define BATCH  2
#define DMODEL 512
#define SCALE  0.125f   // 64^-0.5

typedef __attribute__((ext_vector_type(16))) __bf16 v16bf;
typedef __attribute__((ext_vector_type(8)))  float  v8f;

union ABu { v16bf v; unsigned u[8]; };

__device__ inline v8f wmma_bf16(v16bf a, v16bf b, v8f c) {
  // D = A(16x32) * B(32x16) + C, f32 accumulate
  return __builtin_amdgcn_wmma_f32_16x16x32_bf16(false, a, false, b, (short)0, c, false, false);
}

// A-operand gather (16x32 bf16, ISA 7.12.2): lane row m = lane&15,
// element e -> K = e + 8*((e>>3) + (lane>>4)). Pairs are K-contiguous -> uint loads.
__device__ inline v16bf gather_a(const __bf16* base, int stride_u, int row, int kofs_u, int half) {
  ABu r;
  const unsigned* p = (const unsigned*)base + (size_t)row * stride_u + kofs_u;
#pragma unroll
  for (int i = 0; i < 8; ++i)
    r.u[i] = p[i + ((i >= 4) ? 4 : 0) + (half << 2)];
  return r.v;
}

// B-operand gather (32x16 bf16, ISA 7.12.2 / sparse-B tables): lane col n = lane&15,
// element e -> K = e + 8*((e>>3) + (lane>>4)) (same order as A).
__device__ inline v16bf gather_b(const __bf16* base, int stride_u, int row, int kofs_u, int half) {
  ABu r;
  const unsigned* p = (const unsigned*)base + (size_t)row * stride_u + kofs_u;
#pragma unroll
  for (int i = 0; i < 8; ++i) r.u[i] = p[i + ((i >= 4) ? 4 : 0) + (half << 2)];
  return r.v;
}

// 16-byte async DMA: global -> LDS, tracked by ASYNCcnt (CDNA5 §10 async ops).
// Low 32 bits of a generic LDS pointer are the workgroup-relative LDS offset.
__device__ inline void async_cp16(void* lds, const void* g) {
  unsigned lo = (unsigned)(unsigned long long)lds;
  unsigned long long ga = (unsigned long long)g;
  asm volatile("global_load_async_to_lds_b128 %0, %1, off"
               :: "v"(lo), "v"(ga) : "memory");
}
__device__ inline void wait_async0() { asm volatile("s_wait_asynccnt 0x0" ::: "memory"); }
__device__ inline void wait_async2() { asm volatile("s_wait_asynccnt 0x2" ::: "memory"); }

// ---------------------------------------------------------------------------
// Per-head Cayley orthogonalization. One block per head (grid = 8).
// W head slice Wh (64x512), s = alpha/||Wh||_F.
// A = s(U - U^T) + s^2 V^T V ; M = I + A ; Minv via Gauss-Jordan in LDS.
// Q[i][j<64]   = 2*Minv[j][i] - (i==j)
// Q[i][64+r]   = -2*s * sum_q Minv[q][i] * Wh[q][64+r]
// ---------------------------------------------------------------------------
__global__ __launch_bounds__(256) void cayley_kernel(const float* __restrict__ W,
                                                     const float* __restrict__ alphaP,
                                                     __bf16* __restrict__ Q) {
  const int h = blockIdx.x;
  const int t = threadIdx.x;
  __shared__ float M[64][130];   // [ I+A | I->Minv ], padded
  __shared__ float red[256];
  const float* Wh = W + (size_t)h * 64 * 512;

  // Frobenius norm
  float acc = 0.f;
  for (int i = t; i < 64 * 512; i += 256) { float w = Wh[i]; acc += w * w; }
  red[t] = acc; __syncthreads();
  for (int s2 = 128; s2 > 0; s2 >>= 1) { if (t < s2) red[t] += red[t + s2]; __syncthreads(); }
  const float s = (*alphaP) / sqrtf(red[0]);
  __syncthreads();

  // Build M = I + A and identity in right half
  for (int idx = t; idx < 4096; idx += 256) {
    int i = idx >> 6, j = idx & 63;
    const float* ri = Wh + i * 512 + 64;
    const float* rj = Wh + j * 512 + 64;
    float dot = 0.f;
    for (int r = 0; r < 448; ++r) dot += ri[r] * rj[r];
    float a = s * (Wh[j * 512 + i] - Wh[i * 512 + j]) + s * s * dot;
    M[i][j]      = a + ((i == j) ? 1.f : 0.f);
    M[i][64 + j] = (i == j) ? 1.f : 0.f;
  }
  __syncthreads();

  // Gauss-Jordan (I+A = I + skew + PSD -> well conditioned, no pivoting)
  const int r_ = t >> 2;
  const int cg = (t & 3) * 32;
  for (int p = 0; p < 64; ++p) {
    float rd = 1.0f / M[p][p];
    __syncthreads();
    if (t < 128) M[p][t] *= rd;
    __syncthreads();
    float f = M[r_][p];
    __syncthreads();
    if (r_ != p) {
      for (int c = cg; c < cg + 32; ++c) M[r_][c] -= f * M[p][c];
    }
    __syncthreads();
  }

  // Emit Q head rows (bf16)
  for (int idx = t; idx < 4096; idx += 256) {
    int i = idx >> 6, j = idx & 63;
    float q = 2.f * M[j][64 + i] - ((i == j) ? 1.f : 0.f);
    Q[(size_t)(h * 64 + i) * 512 + j] = (__bf16)q;
  }
  for (int idx = t; idx < 64 * 448; idx += 256) {
    int i = idx / 448, r = idx % 448;
    float a2 = 0.f;
    for (int q2 = 0; q2 < 64; ++q2) a2 += M[q2][64 + i] * Wh[q2 * 512 + 64 + r];
    Q[(size_t)(h * 64 + i) * 512 + 64 + r] = (__bf16)(-2.f * s * a2);
  }
}

// ---------------------------------------------------------------------------
// QKV projection: (8192x512 f32) @ Qw^T -> bf16, per-head layouts.
// Block tile 128x64, 8 waves, each wave = 16 rows x 64 cols (4 WMMA tiles).
// W tile staged via async DMA, x tile converted f32->bf16 (float4 loads).
// z: 0->q [b,h,n,d], 1->k [b,h,n,d], 2->v transposed [b,h,d,n]
// ---------------------------------------------------------------------------
__global__ __launch_bounds__(256) void proj_qkv(const float* __restrict__ x,
                                                const __bf16* __restrict__ Qw,
                                                __bf16* __restrict__ qo,
                                                __bf16* __restrict__ ko,
                                                __bf16* __restrict__ vo) {
  const int z = blockIdx.z;
  const __bf16* Wz = Qw + (size_t)z * DMODEL * DMODEL;
  const int m0 = blockIdx.x * 128;
  const int n0 = blockIdx.y * 64;
  __shared__ __bf16 xs[128 * 32];
  __shared__ __bf16 wsd[64 * 32];
  const int t = threadIdx.x, wave = t >> 5, lane = t & 31, half = lane >> 4, ln = lane & 15;
  const v8f vzero = {0.f, 0.f, 0.f, 0.f, 0.f, 0.f, 0.f, 0.f};
  v8f acc[4] = {vzero, vzero, vzero, vzero};

  for (int k0 = 0; k0 < DMODEL; k0 += 32) {
    // async-DMA the bf16 weight tile (64x32 = 4KB, one b128 per thread)
    {
      int row = t >> 2, seg = (t & 3) * 8;
      async_cp16(wsd + row * 32 + seg,
                 Wz + (size_t)(n0 + row) * DMODEL + k0 + seg);
    }
    // convert-stage x tile (128x32): float4 loads, packed bf16 stores
    for (int i = t; i < 128 * 8; i += 256) {
      int m = i >> 3, kq = (i & 7) * 4;
      float4 vv = *(const float4*)(x + (size_t)(m0 + m) * DMODEL + k0 + kq);
      union { __bf16 h[4]; uint2 u; } pk;
      pk.h[0] = (__bf16)vv.x; pk.h[1] = (__bf16)vv.y;
      pk.h[2] = (__bf16)vv.z; pk.h[3] = (__bf16)vv.w;
      *(uint2*)(xs + m * 32 + kq) = pk.u;
    }
    wait_async0();
    __syncthreads();
    v16bf a = gather_a(xs, 16, wave * 16 + ln, 0, half);
#pragma unroll
    for (int tn = 0; tn < 4; ++tn) {
      v16bf bm = gather_b(wsd, 16, 16 * tn + ln, 0, half);
      acc[tn] = wmma_bf16(a, bm, acc[tn]);
    }
    __syncthreads();
  }

  __bf16* dst = (z == 0) ? qo : (z == 1) ? ko : vo;
  const bool tr = (z == 2);
#pragma unroll
  for (int tn = 0; tn < 4; ++tn)
#pragma unroll
    for (int r = 0; r < 8; ++r) {
      int m = m0 + wave * 16 + r + 8 * half;     // 0..8191
      int n = n0 + 16 * tn + ln;                 // 0..511
      int b = m >> 12, nn = m & 4095, hh = n >> 6, d = n & 63;
      size_t idx = tr ? ((size_t)(b * HEADS + hh) * DIMH + d) * SEQ + nn
                      : ((size_t)(b * HEADS + hh) * SEQ + nn) * DIMH + d;
      dst[idx] = (__bf16)acc[tn][r];
    }
}

// ---------------------------------------------------------------------------
// Flash attention forward. grid = (SEQ/128, BATCH*HEADS), block 256 (8 waves).
// Each wave owns 16 query rows; key blocks of 32; online softmax in f32.
// K/V tiles staged once per workgroup via double-buffered async DMA.
// ---------------------------------------------------------------------------
__global__ __launch_bounds__(256) void flash_attn(const __bf16* __restrict__ qg,
                                                  const __bf16* __restrict__ kg,
                                                  const __bf16* __restrict__ vtg,
                                                  __bf16* __restrict__ og) {
  const int bh = blockIdx.y;
  const int b = bh >> 3, h = bh & 7;
  const int t = threadIdx.x;
  const int wave = t >> 5;
  const int lane = t & 31;
  const int half = lane >> 4, ln = lane & 15;
  const int q0 = blockIdx.x * 128 + wave * 16;
  const __bf16* qh = qg  + (size_t)bh * SEQ * DIMH;
  const __bf16* kh = kg  + (size_t)bh * SEQ * DIMH;
  const __bf16* vh = vtg + (size_t)bh * DIMH * SEQ;   // [d][n]

  __shared__ __bf16 ks[2][32 * 64];   // [key][d]
  __shared__ __bf16 vs[2][64 * 32];   // [d][key]
  __shared__ __bf16 pbuf[8][16 * 32]; // per-wave P scratch
  __bf16* myp = pbuf[wave];

  // staging geometry: one 16B transfer per thread per tile half (K and V)
  const int krow = t >> 3, kseg = (t & 7) * 8;   // 32 rows x 8 segs of 8 elems
  const int vrow = t >> 2, vseg = (t & 3) * 8;   // 64 rows x 4 segs of 8 elems

  v16bf aq0 = gather_a(qh, 32, q0 + ln, 0, half);     // d 0..31
  v16bf aq1 = gather_a(qh, 32, q0 + ln, 16, half);    // d 32..63

  const v8f vzero = {0.f, 0.f, 0.f, 0.f, 0.f, 0.f, 0.f, 0.f};
  v8f o0 = vzero, o1 = vzero, o2 = vzero, o3 = vzero;
  float mi[8], li[8];
#pragma unroll
  for (int r = 0; r < 8; ++r) { mi[r] = -3.0e38f; li[r] = 0.f; }

  const int NB = SEQ / 32;
  // prologue: stage tile 0 into buffer 0
  async_cp16(&ks[0][krow * 64 + kseg], kh + (size_t)krow * DIMH + kseg);
  async_cp16(&vs[0][vrow * 32 + vseg], vh + (size_t)vrow * SEQ + vseg);

  for (int kb = 0; kb < NB; ++kb) {
    const int cur = kb & 1;
    if (kb + 1 < NB) {
      // prefetch next tile into the other buffer, then wait for current tile
      const int key1 = (kb + 1) * 32;
      async_cp16(&ks[cur ^ 1][krow * 64 + kseg], kh + (size_t)(key1 + krow) * DIMH + kseg);
      async_cp16(&vs[cur ^ 1][vrow * 32 + vseg], vh + (size_t)vrow * SEQ + key1 + vseg);
      wait_async2();           // 2 newest outstanding = next tile; current tile done
    } else {
      wait_async0();
    }
    __syncthreads();
    const __bf16* ksb = ks[cur];
    const __bf16* vsb = vs[cur];

    // S = Q K^T : B(k=d, n=key) gathered from staged K rows (d contiguous)
    v8f s0 = vzero, s1 = vzero;
    {
      v16bf b00 = gather_b(ksb, 32, ln,      0,  half);
      v16bf b01 = gather_b(ksb, 32, ln,      16, half);
      v16bf b10 = gather_b(ksb, 32, 16 + ln, 0,  half);
      v16bf b11 = gather_b(ksb, 32, 16 + ln, 16, half);
      s0 = wmma_bf16(aq0, b00, s0);
      s0 = wmma_bf16(aq1, b01, s0);
      s1 = wmma_bf16(aq0, b10, s1);
      s1 = wmma_bf16(aq1, b11, s1);
    }
    // online softmax; C/D layout: vgpr r holds row (r + 8*half), col = ln
    float corr[8];
#pragma unroll
    for (int r = 0; r < 8; ++r) {
      float v0 = s0[r] * SCALE;
      float v1 = s1[r] * SCALE;
      float rm = fmaxf(v0, v1);
#pragma unroll
      for (int msk = 8; msk >= 1; msk >>= 1) rm = fmaxf(rm, __shfl_xor(rm, msk, 16));
      float nm = fmaxf(mi[r], rm);
      float e0 = __expf(v0 - nm);
      float e1 = __expf(v1 - nm);
      float rs = e0 + e1;
#pragma unroll
      for (int msk = 8; msk >= 1; msk >>= 1) rs += __shfl_xor(rs, msk, 16);
      float c = __expf(mi[r] - nm);
      li[r] = li[r] * c + rs;
      mi[r] = nm;
      corr[r] = c;
      s0[r] = e0; s1[r] = e1;
    }
#pragma unroll
    for (int r = 0; r < 8; ++r) {
      o0[r] *= corr[r]; o1[r] *= corr[r]; o2[r] *= corr[r]; o3[r] *= corr[r];
    }
    // P: C/D layout -> row-major 16x32 in per-wave LDS, re-gather as A operand
#pragma unroll
    for (int r = 0; r < 8; ++r) {
      int m = r + 8 * half;
      myp[m * 32 + ln]      = (__bf16)s0[r];
      myp[m * 32 + 16 + ln] = (__bf16)s1[r];
    }
    asm volatile("s_wait_dscnt 0x0" ::: "memory");
    v16bf ap = gather_a(myp, 16, ln, 0, half);
    // O += P V : B(k=key, n=d) from staged transposed V rows (keys contiguous)
    {
      v16bf bv0 = gather_b(vsb, 16, 0  + ln, 0, half);
      v16bf bv1 = gather_b(vsb, 16, 16 + ln, 0, half);
      v16bf bv2 = gather_b(vsb, 16, 32 + ln, 0, half);
      v16bf bv3 = gather_b(vsb, 16, 48 + ln, 0, half);
      o0 = wmma_bf16(ap, bv0, o0);
      o1 = wmma_bf16(ap, bv1, o1);
      o2 = wmma_bf16(ap, bv2, o2);
      o3 = wmma_bf16(ap, bv3, o3);
    }
    __syncthreads();   // protect ks/vs[cur^1]... reuse next iteration
  }

  // normalize and write [b, n, h*64 + d] bf16
#pragma unroll
  for (int r = 0; r < 8; ++r) {
    int row = q0 + r + 8 * half;
    float inv = 1.f / li[r];
    size_t ob = ((size_t)(b * SEQ + row)) * DMODEL + h * DIMH + ln;
    og[ob +  0] = (__bf16)(o0[r] * inv);
    og[ob + 16] = (__bf16)(o1[r] * inv);
    og[ob + 32] = (__bf16)(o2[r] * inv);
    og[ob + 48] = (__bf16)(o3[r] * inv);
  }
}

// ---------------------------------------------------------------------------
// Output projection: attn(8192x512 bf16) @ Qo^T + bo -> f32 d_out
// Both tiles are bf16 -> staged entirely via async DMA.
// ---------------------------------------------------------------------------
__global__ __launch_bounds__(256) void proj_out(const __bf16* __restrict__ ain,
                                                const __bf16* __restrict__ Wo,
                                                const float* __restrict__ bo,
                                                float* __restrict__ out) {
  const int m0 = blockIdx.x * 128;
  const int n0 = blockIdx.y * 64;
  __shared__ __bf16 as_[128 * 32];
  __shared__ __bf16 wsd[64 * 32];
  const int t = threadIdx.x, wave = t >> 5, lane = t & 31, half = lane >> 4, ln = lane & 15;
  const v8f vzero = {0.f, 0.f, 0.f, 0.f, 0.f, 0.f, 0.f, 0.f};
  v8f acc[4] = {vzero, vzero, vzero, vzero};

  for (int k0 = 0; k0 < DMODEL; k0 += 32) {
    // A tile: 128x32 bf16 = 8KB -> two b128 per thread
#pragma unroll
    for (int half_i = 0; half_i < 2; ++half_i) {
      int i = t + half_i * 256;
      int row = i >> 2, seg = (i & 3) * 8;
      async_cp16(as_ + row * 32 + seg,
                 ain + (size_t)(m0 + row) * DMODEL + k0 + seg);
    }
    // W tile: 64x32 bf16 = 4KB -> one b128 per thread
    {
      int row = t >> 2, seg = (t & 3) * 8;
      async_cp16(wsd + row * 32 + seg,
                 Wo + (size_t)(n0 + row) * DMODEL + k0 + seg);
    }
    wait_async0();
    __syncthreads();
    v16bf a = gather_a(as_, 16, wave * 16 + ln, 0, half);
#pragma unroll
    for (int tn = 0; tn < 4; ++tn) {
      v16bf bm = gather_b(wsd, 16, 16 * tn + ln, 0, half);
      acc[tn] = wmma_bf16(a, bm, acc[tn]);
    }
    __syncthreads();
  }

#pragma unroll
  for (int tn = 0; tn < 4; ++tn)
#pragma unroll
    for (int r = 0; r < 8; ++r) {
      int m = m0 + wave * 16 + r + 8 * half;
      int n = n0 + 16 * tn + ln;
      out[(size_t)m * DMODEL + n] = acc[tn][r] + bo[n];
    }
}

// ---------------------------------------------------------------------------
extern "C" void kernel_launch(void* const* d_in, const int* in_sizes, int n_in,
                              void* d_out, int out_size, void* d_ws, size_t ws_size,
                              hipStream_t stream) {
  (void)in_sizes; (void)n_in; (void)out_size; (void)ws_size;
  const float* x   = (const float*)d_in[0];
  const float* Wq  = (const float*)d_in[1];
  const float* aq  = (const float*)d_in[2];
  const float* Wk  = (const float*)d_in[3];
  const float* ak  = (const float*)d_in[4];
  const float* Wv  = (const float*)d_in[5];
  const float* av  = (const float*)d_in[6];
  const float* Wo  = (const float*)d_in[7];
  const float* ao  = (const float*)d_in[8];
  const float* bo  = (const float*)d_in[9];
  float* out = (float*)d_out;

  // workspace layout (bf16 elements)
  const size_t QW   = (size_t)DMODEL * DMODEL;                 // 262144
  const size_t QKV  = (size_t)BATCH * HEADS * SEQ * DIMH;      // 4194304
  __bf16* Qw   = (__bf16*)d_ws;            // 4 matrices: q,k,v,o
  __bf16* qb   = Qw + 4 * QW;
  __bf16* kb   = qb + QKV;
  __bf16* vb   = kb + QKV;                 // stored transposed [b,h,d,n]
  __bf16* attn = vb + QKV;                 // [b,n,512]

  cayley_kernel<<<HEADS, 256, 0, stream>>>(Wq, aq, Qw + 0 * QW);
  cayley_kernel<<<HEADS, 256, 0, stream>>>(Wk, ak, Qw + 1 * QW);
  cayley_kernel<<<HEADS, 256, 0, stream>>>(Wv, av, Qw + 2 * QW);
  cayley_kernel<<<HEADS, 256, 0, stream>>>(Wo, ao, Qw + 3 * QW);

  proj_qkv<<<dim3(BATCH * SEQ / 128, DMODEL / 64, 3), 256, 0, stream>>>(x, Qw, qb, kb, vb);
  flash_attn<<<dim3(SEQ / 128, BATCH * HEADS), 256, 0, stream>>>(qb, kb, vb, attn);
  proj_out<<<dim3(BATCH * SEQ / 128, DMODEL / 64), 256, 0, stream>>>(attn, Qw + 3 * QW, bo, out);
}
